// PaperAlignedAttention_53120155517055
// MI455X (gfx1250) — matrix operand order repaired and checked by reading, not compiled, not run on hardware
//
#include <hip/hip_runtime.h>

// ---------------------------------------------------------------------------
// Shapes (fixed by the reference)
//   B=64 N=197 D=768  H=12 hd=64  E=8 R=8
//   BN = 12608, 3D = 2304, KEXT = 768+64 = 832
// Pipeline:
//   1. prep_weights : Bext[2304][832]=bf16(qkv_w | lora_b^T), projWb=bf16(proj_w)
//   2. build_aext   : Aext[12608][832]=bf16(tokens | gate_e*h_{e,r})
//   3. gemm (bf16 out): qkvb = Aext*Bext^T + qkv_b   (fused qkv + LoRA delta)
//   4. transpose_v  : vT[bh][64][224] = V^T (zero padded keys)
//   5. attn         : attnOut[12608][768] (bf16), per-(b,h) WMMA attention
//   6. gemm (f32 out): d_out = attnOut * projWb^T + proj_b
// ---------------------------------------------------------------------------

typedef __bf16 bf16_t;
typedef __attribute__((ext_vector_type(16))) __bf16 v16bf;
typedef __attribute__((ext_vector_type(8)))  float  v8f;

union ABFrag { v16bf v; uint4 u[2]; };
union AccFrag { v8f v; float f[8]; };

#define BN    12608
#define DMODEL 768
#define D3    2304
#define KEXT  832
#define NKEY_PAD 224   // 197 keys padded to 7 k-steps of 32
#define NROW_PAD 208   // 13 tiles of 16

// ---- CDNA5 async global->LDS staging (guarded: falls back to reg staging) ---
#if defined(__HIP_DEVICE_COMPILE__) && __has_builtin(__builtin_amdgcn_global_load_async_to_lds_b128)
#define USE_ASYNC_LDS 1
typedef int v4i_vec __attribute__((vector_size(16)));
typedef __attribute__((address_space(1))) v4i_vec* gv4i_p;  // global v4i*
typedef __attribute__((address_space(3))) v4i_vec* lv4i_p;  // LDS v4i*

static __device__ __forceinline__ void async_copy_b128(const void* g, void* l)
{
    __builtin_amdgcn_global_load_async_to_lds_b128((gv4i_p)g, (lv4i_p)l, 0, 0);
}
#else
#define USE_ASYNC_LDS 0
#endif

static __device__ __forceinline__ void wait_async0()
{
#if defined(__HIP_DEVICE_COMPILE__)
#if __has_builtin(__builtin_amdgcn_s_wait_asynccnt)
    __builtin_amdgcn_s_wait_asynccnt(0);
#else
    asm volatile("s_wait_asynccnt 0x0" ::: "memory");
#endif
#endif
}

// ---------------------------------------------------------------------------
// Generic bf16 GEMM:  C[M][Nc] = A[M][K] * Bm[Nc][K]^T + bias
// Both operands K-contiguous. Block: 256 thr = 8 waves; block tile 128x128;
// wave tile 64x32 (8 WMMAs / 32-deep k-step). Double-buffered LDS; the next
// tile streams global->LDS via the async path (ASYNCcnt) while WMMAs run on
// the current one.
// ---------------------------------------------------------------------------
__global__ __launch_bounds__(256)
void gemm_bf16_kernel(const bf16_t* __restrict__ A, const bf16_t* __restrict__ Bm,
                      const float* __restrict__ bias, float* __restrict__ Cf,
                      bf16_t* __restrict__ Cb, int M, int Nc, int K)
{
    __shared__ bf16_t As[2][128 * 32];
    __shared__ bf16_t Bs[2][128 * 32];

    const int tid  = threadIdx.x;
    const int lane = tid & 31;
    const int wave = tid >> 5;
    const int r    = lane & 15;   // M-row / N-col within subtile
    const int kh   = lane >> 4;   // K-half selector
    const int wm   = (wave & 1) * 64;   // wave M offset within block tile
    const int wn   = (wave >> 1) * 32;  // wave N offset within block tile
    const int m0   = blockIdx.y * 128;
    const int n0   = blockIdx.x * 128;

    // Staging addresses for this thread (2 chunks of 8 bf16 per tile each).
    // A rows beyond M are CLAMPED (not zeroed): an A-tile row feeds only the
    // same C row, which the epilogue masks, so no zero-fill and no exec-mask
    // branches are needed.
    int rowA0 = tid >> 2, kcA0 = tid & 3;
    int rowA1 = (tid + 256) >> 2, kcA1 = tid & 3;
    int grA0 = m0 + rowA0; if (grA0 >= M) grA0 = M - 1;
    int grA1 = m0 + rowA1; if (grA1 >= M) grA1 = M - 1;

    auto stage = [&](int buf, int kt) {
        const bf16_t* gA0 = A + (size_t)grA0 * K + kt + kcA0 * 8;
        const bf16_t* gA1 = A + (size_t)grA1 * K + kt + kcA1 * 8;
        const bf16_t* gB0 = Bm + (size_t)(n0 + rowA0) * K + kt + kcA0 * 8;
        const bf16_t* gB1 = Bm + (size_t)(n0 + rowA1) * K + kt + kcA1 * 8;
        bf16_t* lA0 = &As[buf][rowA0 * 32 + kcA0 * 8];
        bf16_t* lA1 = &As[buf][rowA1 * 32 + kcA1 * 8];
        bf16_t* lB0 = &Bs[buf][rowA0 * 32 + kcA0 * 8];
        bf16_t* lB1 = &Bs[buf][rowA1 * 32 + kcA1 * 8];
#if USE_ASYNC_LDS
        async_copy_b128(gA0, lA0);
        async_copy_b128(gA1, lA1);
        async_copy_b128(gB0, lB0);
        async_copy_b128(gB1, lB1);
#else
        uint4 a0 = *(const uint4*)gA0;
        uint4 a1 = *(const uint4*)gA1;
        uint4 b0 = *(const uint4*)gB0;
        uint4 b1 = *(const uint4*)gB1;
        *(uint4*)lA0 = a0;
        *(uint4*)lA1 = a1;
        *(uint4*)lB0 = b0;
        *(uint4*)lB1 = b1;
#endif
    };

    AccFrag acc[4][2];
#pragma unroll
    for (int i = 0; i < 4; ++i)
#pragma unroll
        for (int j = 0; j < 2; ++j)
#pragma unroll
            for (int q = 0; q < 8; ++q) acc[i][j].f[q] = 0.0f;

    const int kTiles = K >> 5;  // K % 32 == 0 for all our GEMMs

    stage(0, 0);
#if USE_ASYNC_LDS
    wait_async0();
#endif
    __syncthreads();

    for (int t = 0; t < kTiles; ++t) {
        const int cur = t & 1;
        if (t + 1 < kTiles) stage(1 - cur, (t + 1) * 32);  // overlap with WMMAs

        // ---- fragments (ISA 16-bit A/B layouts) ----
        ABFrag a[4], b[2];
#pragma unroll
        for (int i = 0; i < 4; ++i) {
            const bf16_t* p = &As[cur][(wm + i * 16 + r) * 32 + kh * 8];
            a[i].u[0] = *(const uint4*)(p);        // K = kh*8 .. +7
            a[i].u[1] = *(const uint4*)(p + 16);   // K = 16+kh*8 .. +7
        }
#pragma unroll
        for (int j = 0; j < 2; ++j) {
            const bf16_t* p = &Bs[cur][(wn + j * 16 + r) * 32 + kh * 8];
            b[j].u[0] = *(const uint4*)(p);        // K = kh*8 .. +7
            b[j].u[1] = *(const uint4*)(p + 16);   // K = 16+kh*8 .. +7
        }
#pragma unroll
        for (int i = 0; i < 4; ++i)
#pragma unroll
            for (int j = 0; j < 2; ++j)
                acc[i][j].v = __builtin_amdgcn_wmma_f32_16x16x32_bf16(
                    false, a[i].v, false, b[j].v, (short)0, acc[i][j].v, false, false);

#if USE_ASYNC_LDS
        wait_async0();          // this wave's async writes into next buffer done
#endif
        __syncthreads();        // all waves: next buffer ready / current consumed
    }

    // ---- epilogue: bias + store (C layout: VGPR q -> row q + kh*8; col = r)
#pragma unroll
    for (int i = 0; i < 4; ++i)
#pragma unroll
        for (int j = 0; j < 2; ++j)
#pragma unroll
            for (int q = 0; q < 8; ++q) {
                int row = m0 + wm + i * 16 + kh * 8 + q;
                int col = n0 + wn + j * 16 + r;
                if (row < M) {
                    float v = acc[i][j].f[q] + (bias ? bias[col] : 0.0f);
                    if (Cb) Cb[(size_t)row * Nc + col] = (bf16_t)v;
                    else    Cf[(size_t)row * Nc + col] = v;
                }
            }
}

// ---------------------------------------------------------------------------
// Weight prep: Bext[n][k] = k<768 ? qkv_w[n][k] : lora_b[k-768][n];  projWb=bf16(proj_w)
// ---------------------------------------------------------------------------
__global__ void prep_weights_kernel(const float* __restrict__ qkv_w,
                                    const float* __restrict__ lora_b,
                                    const float* __restrict__ proj_w,
                                    bf16_t* __restrict__ Bext,
                                    bf16_t* __restrict__ projWb)
{
    const size_t total1 = (size_t)D3 * KEXT;
    const size_t total2 = (size_t)DMODEL * DMODEL;
    for (size_t idx = (size_t)blockIdx.x * blockDim.x + threadIdx.x;
         idx < total1 + total2; idx += (size_t)gridDim.x * blockDim.x) {
        if (idx < total1) {
            int n = (int)(idx / KEXT), k = (int)(idx % KEXT);
            float v = (k < DMODEL) ? qkv_w[(size_t)n * DMODEL + k]
                                   : lora_b[(size_t)(k - DMODEL) * D3 + n];
            Bext[idx] = (bf16_t)v;
        } else {
            size_t i2 = idx - total1;
            projWb[i2] = (bf16_t)proj_w[i2];
        }
    }
}

// ---------------------------------------------------------------------------
// Per-token: Aext = [bf16(token) | gate_e * h_{e,r}] ; one block per token.
// ---------------------------------------------------------------------------
__global__ __launch_bounds__(256)
void build_aext_kernel(const float* __restrict__ tokens,
                       const float* __restrict__ gate_w,
                       const float* __restrict__ lora_a,
                       bf16_t* __restrict__ Aext)
{
    __shared__ float tok[DMODEL];
    __shared__ float hv[72];  // [0..7]=gate logits, [8..71]=h[e][r]
    const int t = blockIdx.x, tid = threadIdx.x;
    const float* tp = tokens + (size_t)t * DMODEL;
    bf16_t* arow = Aext + (size_t)t * KEXT;

    for (int i = tid; i < DMODEL; i += 256) {
        float v = tp[i];
        tok[i] = v;
        arow[i] = (bf16_t)v;
    }
    __syncthreads();

    if (tid < 72) {
        float acc = 0.0f;
        if (tid < 8) {
            const float* w = gate_w + (size_t)tid * DMODEL;
            for (int d = 0; d < DMODEL; ++d) acc += tok[d] * w[d];
        } else {
            int er = tid - 8, e = er >> 3, rr = er & 7;
            const float* la = lora_a + (size_t)e * DMODEL * 8 + rr;
            for (int d = 0; d < DMODEL; ++d) acc += tok[d] * la[(size_t)d * 8];
        }
        hv[tid] = acc;
    }
    __syncthreads();

    if (tid < 64) {
        float mx = -1e30f;
        for (int k = 0; k < 8; ++k) mx = fmaxf(mx, hv[k]);
        float s = 0.0f;
        for (int k = 0; k < 8; ++k) s += __expf(hv[k] - mx);
        int e = tid >> 3;
        float g = __expf(hv[e] - mx) / s;
        arow[DMODEL + tid] = (bf16_t)(g * hv[8 + tid]);
    }
}

// ---------------------------------------------------------------------------
// vT[bh][d][n] = V[b,h,n,d], zero for n >= 197 (keys padded to 224)
// ---------------------------------------------------------------------------
__global__ void transpose_v_kernel(const bf16_t* __restrict__ qkvb,
                                   bf16_t* __restrict__ vT)
{
    const size_t total = (size_t)768 * 64 * NKEY_PAD;
    for (size_t idx = (size_t)blockIdx.x * blockDim.x + threadIdx.x;
         idx < total; idx += (size_t)gridDim.x * blockDim.x) {
        int n = (int)(idx % NKEY_PAD);
        size_t rem = idx / NKEY_PAD;
        int d = (int)(rem & 63);
        int bh = (int)(rem >> 6);
        int b = bh / 12, h = bh % 12;
        bf16_t v = (bf16_t)0.0f;
        if (n < 197)
            v = qkvb[(size_t)(b * 197 + n) * D3 + 2 * DMODEL + h * 64 + d];
        vT[idx] = v;
    }
}

// ---------------------------------------------------------------------------
// Attention: one block per (b,h); 8 waves, each owns 16-query row tiles.
// S = Q K^T (bf16 WMMA, f32 acc) -> masked softmax (shfl_xor width-16) ->
// P staged per-wave in LDS -> O = P V (WMMA vs vT) -> attnOut bf16.
// ---------------------------------------------------------------------------
__global__ __launch_bounds__(256)
void attn_kernel(const bf16_t* __restrict__ qkvb, const bf16_t* __restrict__ vT,
                 bf16_t* __restrict__ attnOut)
{
    __shared__ bf16_t P[8][16 * NKEY_PAD];   // 57344 B

    const int tid = threadIdx.x, lane = tid & 31, wave = tid >> 5;
    const int r = lane & 15, kh = lane >> 4;
    const int bh = blockIdx.x, b = bh / 12, h = bh % 12;

    const bf16_t* qb = qkvb + (size_t)(b * 197) * D3 + h * 64;
    const bf16_t* kb = qb + DMODEL;
    const bf16_t* vb = vT + (size_t)bh * 64 * NKEY_PAD;

    for (int qt = wave; qt < 13; qt += 8) {
        // prefetch next q tile's rows for this wave (global_prefetch_b8)
        {
            int nq = qt + 8;
            if (nq < 13) {
                int prow = nq * 16 + r; if (prow > 196) prow = 196;
                __builtin_prefetch(qb + (size_t)prow * D3, 0, 3);
            }
        }

        int qrow = qt * 16 + r; if (qrow > 196) qrow = 196;  // clamped rows never stored
        const bf16_t* qp = qb + (size_t)qrow * D3;
        ABFrag aq[2];
#pragma unroll
        for (int ks = 0; ks < 2; ++ks) {
            aq[ks].u[0] = *(const uint4*)(qp + ks * 32 + kh * 8);
            aq[ks].u[1] = *(const uint4*)(qp + ks * 32 + 16 + kh * 8);
        }

        // ---- S = Q K^T : 13 key tiles x 2 k-steps ----
        AccFrag s[13];
#pragma unroll
        for (int j = 0; j < 13; ++j)
#pragma unroll
            for (int q = 0; q < 8; ++q) s[j].f[q] = 0.0f;

#pragma unroll
        for (int j = 0; j < 13; ++j) {
            int krow = j * 16 + r; if (krow > 196) krow = 196;
            const bf16_t* kp = kb + (size_t)krow * D3;
#pragma unroll
            for (int ks = 0; ks < 2; ++ks) {
                ABFrag bk;
                bk.u[0] = *(const uint4*)(kp + ks * 32 + kh * 8);
                bk.u[1] = *(const uint4*)(kp + ks * 32 + kh * 8 + 16);
                s[j].v = __builtin_amdgcn_wmma_f32_16x16x32_bf16(
                    false, aq[ks].v, false, bk.v, (short)0, s[j].v, false, false);
            }
        }

        // ---- scale + key mask ----
#pragma unroll
        for (int j = 0; j < 13; ++j) {
            bool oob = (j * 16 + r) > 196;
#pragma unroll
            for (int q = 0; q < 8; ++q) {
                float t = s[j].f[q] * 0.125f;   // hd^-0.5, hd=64
                s[j].f[q] = oob ? -1e30f : t;
            }
        }

        // ---- row softmax (rows live across 16-lane groups) ----
        float ls[8];
#pragma unroll
        for (int q = 0; q < 8; ++q) {
            float m = -1e30f;
#pragma unroll
            for (int j = 0; j < 13; ++j) m = fmaxf(m, s[j].f[q]);
            m = fmaxf(m, __shfl_xor(m, 1, 16));
            m = fmaxf(m, __shfl_xor(m, 2, 16));
            m = fmaxf(m, __shfl_xor(m, 4, 16));
            m = fmaxf(m, __shfl_xor(m, 8, 16));
            float sum = 0.0f;
#pragma unroll
            for (int j = 0; j < 13; ++j) {
                float p = __expf(s[j].f[q] - m);
                s[j].f[q] = p;
                sum += p;
            }
            sum += __shfl_xor(sum, 1, 16);
            sum += __shfl_xor(sum, 2, 16);
            sum += __shfl_xor(sum, 4, 16);
            sum += __shfl_xor(sum, 8, 16);
            ls[q] = sum;
        }

        // ---- P to per-wave LDS (bf16), zero-pad cols 208..223 ----
        bf16_t* pw = &P[wave][0];
#pragma unroll
        for (int q = 0; q < 8; ++q)
            pw[(kh * 8 + q) * NKEY_PAD + NROW_PAD + r] = (bf16_t)0.0f;
#pragma unroll
        for (int j = 0; j < 13; ++j)
#pragma unroll
            for (int q = 0; q < 8; ++q)
                pw[(kh * 8 + q) * NKEY_PAD + j * 16 + r] = (bf16_t)s[j].f[q];

        // ---- O = P V : 7 k-steps x 4 hd tiles ----
        AccFrag o[4];
#pragma unroll
        for (int j2 = 0; j2 < 4; ++j2)
#pragma unroll
            for (int q = 0; q < 8; ++q) o[j2].f[q] = 0.0f;

#pragma unroll
        for (int ks = 0; ks < 7; ++ks) {
            ABFrag ap;
            const bf16_t* pp = pw + r * NKEY_PAD + ks * 32 + kh * 8;
            ap.u[0] = *(const uint4*)(pp);
            ap.u[1] = *(const uint4*)(pp + 16);
#pragma unroll
            for (int j2 = 0; j2 < 4; ++j2) {
                ABFrag bv;
                const bf16_t* vp = vb + (size_t)(j2 * 16 + r) * NKEY_PAD + ks * 32 + kh * 8;
                bv.u[0] = *(const uint4*)(vp);
                bv.u[1] = *(const uint4*)(vp + 16);
                o[j2].v = __builtin_amdgcn_wmma_f32_16x16x32_bf16(
                    false, ap.v, false, bv.v, (short)0, o[j2].v, false, false);
            }
        }

        // ---- normalize + store ----
#pragma unroll
        for (int j2 = 0; j2 < 4; ++j2)
#pragma unroll
            for (int q = 0; q < 8; ++q) {
                int m = kh * 8 + q;
                int n = qt * 16 + m;
                if (n < 197)
                    attnOut[(size_t)(b * 197 + n) * DMODEL + h * 64 + j2 * 16 + r] =
                        (bf16_t)(o[j2].f[q] / ls[q]);
            }
    }
}

// ---------------------------------------------------------------------------
extern "C" void kernel_launch(void* const* d_in, const int* in_sizes, int n_in,
                              void* d_out, int out_size, void* d_ws, size_t ws_size,
                              hipStream_t stream)
{
    const float* tokens = (const float*)d_in[0];
    const float* qkv_w  = (const float*)d_in[1];
    const float* qkv_b  = (const float*)d_in[2];
    const float* proj_w = (const float*)d_in[3];
    const float* proj_b = (const float*)d_in[4];
    const float* gate_w = (const float*)d_in[5];
    const float* lora_a = (const float*)d_in[6];
    const float* lora_b = (const float*)d_in[7];
    float* out = (float*)d_out;

    // workspace carve (~124 MB), 256B aligned
    char* ws = (char*)d_ws;
    auto alloc = [&](size_t bytes) -> char* {
        char* p = ws;
        ws += (bytes + 255) & ~(size_t)255;
        return p;
    };
    bf16_t* Aext    = (bf16_t*)alloc((size_t)BN * KEXT * 2);      // 21.0 MB
    bf16_t* Bext    = (bf16_t*)alloc((size_t)D3 * KEXT * 2);      //  3.8 MB
    bf16_t* projWb  = (bf16_t*)alloc((size_t)DMODEL * DMODEL * 2);//  1.2 MB
    bf16_t* qkvb    = (bf16_t*)alloc((size_t)BN * D3 * 2);        // 58.1 MB
    bf16_t* vT      = (bf16_t*)alloc((size_t)768 * 64 * NKEY_PAD * 2); // 22.0 MB
    bf16_t* attnOut = (bf16_t*)alloc((size_t)BN * DMODEL * 2);    // 19.4 MB

    // 1. weight conversion / packing
    prep_weights_kernel<<<4096, 256, 0, stream>>>(qkv_w, lora_b, proj_w, Bext, projWb);

    // 2. token conversion + LoRA routing coefficients
    build_aext_kernel<<<BN, 256, 0, stream>>>(tokens, gate_w, lora_a, Aext);

    // 3. fused QKV + LoRA-delta GEMM: [12608 x 832] x [2304 x 832]^T + qkv_b -> bf16
    {
        dim3 grid(D3 / 128, (BN + 127) / 128);
        gemm_bf16_kernel<<<grid, 256, 0, stream>>>(Aext, Bext, qkv_b,
                                                   nullptr, qkvb, BN, D3, KEXT);
    }

    // 4. V transpose (B-fragment friendly layout, zero-padded keys)
    transpose_v_kernel<<<8192, 256, 0, stream>>>(qkvb, vT);

    // 5. attention per (b,h)
    attn_kernel<<<64 * 12, 256, 0, stream>>>(qkvb, vT, attnOut);

    // 6. output projection: [12608 x 768] x [768 x 768]^T + proj_b -> f32 d_out
    {
        dim3 grid(DMODEL / 128, (BN + 127) / 128);
        gemm_bf16_kernel<<<grid, 256, 0, stream>>>(attnOut, projWb, proj_b,
                                                   out, nullptr, BN, DMODEL, DMODEL);
    }

    (void)in_sizes; (void)n_in; (void)out_size; (void)ws_size;
}
